// BERTSelfAttention_40888088658354
// MI455X (gfx1250) — compile-verified, matrix operand order checked by reading, not run on hardware
//
#include <hip/hip_runtime.h>
#include <hip/hip_bf16.h>

typedef __attribute__((ext_vector_type(16))) _Float16 v16h;
typedef __attribute__((ext_vector_type(8)))  float    v8f;
typedef __attribute__((ext_vector_type(4)))  int      v4i;

#define B_  8
#define S_  1024
#define D_  768
#define H_  12
#define DH_ 64

// ---- CDNA5 async global->LDS copy (ASYNCcnt path), with safe fallback -----
#if defined(__has_builtin)
#  if __has_builtin(__builtin_amdgcn_global_load_async_to_lds_b128)
#    define ASYNC_LDS 1
#  endif
#  if __has_builtin(__builtin_amdgcn_s_wait_asynccnt)
#    define HAVE_WAIT_ASYNC 1
#  endif
#endif

__device__ __forceinline__ void copy16_g2l(void* l, const void* g) {
#ifdef ASYNC_LDS
  __builtin_amdgcn_global_load_async_to_lds_b128(
      (__attribute__((address_space(1))) v4i*)(v4i*)(void*)g,
      (__attribute__((address_space(3))) v4i*)(v4i*)l, 0, 0);
#else
  *(uint4*)l = *(const uint4*)g;
#endif
}

__device__ __forceinline__ void wait_async_copies() {
#ifdef ASYNC_LDS
#  ifdef HAVE_WAIT_ASYNC
  __builtin_amdgcn_s_wait_asynccnt(0);
#  else
  asm volatile("s_wait_asynccnt 0" ::: "memory");
#  endif
#endif
}

// ---------------------------------------------------------------------------
// Kernel 1: fp32 -> f16 conversion of x and W
// ---------------------------------------------------------------------------
__global__ __launch_bounds__(256)
void cvt_kernel(const float* __restrict__ x, const float* __restrict__ W,
                _Float16* __restrict__ xh, _Float16* __restrict__ wh) {
  size_t i = (size_t)blockIdx.x * blockDim.x + threadIdx.x;
  const size_t NX = (size_t)B_ * S_ * D_;
  const size_t NW = (size_t)D_ * D_;
  if (i < NX) xh[i] = (_Float16)x[i];
  if (i < NW) wh[i] = (_Float16)W[i];
}

// ---------------------------------------------------------------------------
// Kernel 2: mixed = x @ W^T + b -> [B,H,S,DH] f16.
// Wave tile 32x64: 2 A fragments share 4 B fragments -> 8 WMMA per K-step,
// two independent accumulator chains for latency hiding.
// ---------------------------------------------------------------------------
__global__ __launch_bounds__(256)
void proj_kernel(const _Float16* __restrict__ xh, const _Float16* __restrict__ wh,
                 const float* __restrict__ bias, _Float16* __restrict__ qh) {
  const int lane = threadIdx.x & 31;
  const int wave = threadIdx.x >> 5;
  const int m0 = blockIdx.x * 256 + wave * 32;   // 32 output rows per wave
  const int n0 = blockIdx.y * 64;                // 64 output cols per wave
  const int lr = lane & 15;
  const int lh = lane >> 4;

  v8f acc[2][4] = {};
  const _Float16* aBase0 = xh + (size_t)(m0 + lr) * D_ + lh * 16;
  const _Float16* aBase1 = xh + (size_t)(m0 + 16 + lr) * D_ + lh * 16;

  for (int k0 = 0; k0 < D_; k0 += 32) {
    v16h a0 = *(const v16h*)(aBase0 + k0);
    v16h a1 = *(const v16h*)(aBase1 + k0);
#pragma unroll
    for (int nt = 0; nt < 4; ++nt) {
      const _Float16* bp = wh + (size_t)(n0 + nt * 16 + lr) * D_ + k0 + lh * 16;
      v16h bfrag = *(const v16h*)bp;
      acc[0][nt] = __builtin_amdgcn_wmma_f32_16x16x32_f16(
          false, a0, false, bfrag, (short)0, acc[0][nt], false, false);
      acc[1][nt] = __builtin_amdgcn_wmma_f32_16x16x32_f16(
          false, a1, false, bfrag, (short)0, acc[1][nt], false, false);
    }
  }

  // Epilogue: bias + scatter to per-head layout [B,H,S,DH] (f16).
#pragma unroll
  for (int rt = 0; rt < 2; ++rt) {
#pragma unroll
    for (int nt = 0; nt < 4; ++nt) {
      int n = n0 + nt * 16 + lr;
      float bv = bias[n];
      int h = n / DH_, d = n % DH_;
#pragma unroll
      for (int r = 0; r < 8; ++r) {
        int m = m0 + rt * 16 + r + lh * 8;   // C layout: row = r + 8*(lane/16)
        int bb = m / S_, s = m % S_;
        qh[(((size_t)bb * H_ + h) * S_ + s) * DH_ + d] =
            (_Float16)(acc[rt][nt][r] + bv);
      }
    }
  }
}

// ---------------------------------------------------------------------------
// Kernel 3: flash attention per (b,h). Q = K = V = qh[bh] (1024x64 f16).
// K tile staged via async global->LDS (ASYNCcnt); V transposed manually.
// ---------------------------------------------------------------------------
__global__ __launch_bounds__(256)
void attn_kernel(const _Float16* __restrict__ qh, const float* __restrict__ mask,
                 float* __restrict__ out) {
  __shared__ _Float16 ldsK[32 * 64];     // [t][d] row-major
  __shared__ _Float16 ldsVT[64 * 32];    // [d][t] transposed
  __shared__ _Float16 ldsP[8][16 * 32];  // per-wave P scratch (C->A relayout)

  const int lane = threadIdx.x & 31;
  const int wave = threadIdx.x >> 5;
  const int bh = blockIdx.y;             // 0..95
  const int b  = bh / H_;
  const int h  = bh % H_;
  const int q0 = blockIdx.x * 128 + wave * 16;
  const int lr = lane & 15;
  const int lh = lane >> 4;

  const _Float16* base = qh + (size_t)bh * S_ * DH_;

  // Q fragments for this wave's 16 rows (d split 0..31 / 32..63)
  v16h qa[2];
#pragma unroll
  for (int kk = 0; kk < 2; ++kk)
    qa[kk] = *(const v16h*)(base + (size_t)(q0 + lr) * DH_ + kk * 32 + lh * 16);

  v8f o[4] = {};
  float mrun[8], lrun[8];
#pragma unroll
  for (int r = 0; r < 8; ++r) { mrun[r] = -3.0e38f; lrun[r] = 0.f; }
  const float scale = 0.125f;  // 1/sqrt(64)

  for (int t0 = 0; t0 < S_; t0 += 32) {
    __syncthreads();  // WAR: prior iteration done reading tiles
    {
      // K tile 32x64 f16 = 4 KB: 256 lanes x 16 B async copies (no VGPR hop)
      copy16_g2l(&ldsK[threadIdx.x * 8], base + (size_t)t0 * DH_ + threadIdx.x * 8);
      // Prefetch next tile toward L2/WGP while async copies run
      if (t0 + 32 < S_)
        __builtin_prefetch(base + (size_t)(t0 + 32) * DH_ + threadIdx.x * 8, 0, 1);
      // V^T: thread handles row tt = tid/8, 8 d's; scalar transpose stores
      int tt = threadIdx.x >> 3;
      int d0 = (threadIdx.x & 7) * 8;
      const _Float16* vrow = base + (size_t)(t0 + tt) * DH_ + d0;
#pragma unroll
      for (int j = 0; j < 8; ++j)
        ldsVT[(d0 + j) * 32 + tt] = vrow[j];
      wait_async_copies();  // own ASYNCcnt drained before the barrier
    }
    __syncthreads();

    // scores: two 16x16 tiles over t = t0..t0+31
    v8f sc[2] = {};
#pragma unroll
    for (int ts = 0; ts < 2; ++ts) {
#pragma unroll
      for (int kk = 0; kk < 2; ++kk) {
        v16h kb = *(const v16h*)(&ldsK[(ts * 16 + lr) * 64 + kk * 32 + lh * 16]);
        sc[ts] = __builtin_amdgcn_wmma_f32_16x16x32_f16(
            false, qa[kk], false, kb, (short)0, sc[ts], false, false);
      }
    }

    float mk0 = mask[(size_t)b * S_ + t0 + lr];
    float mk1 = mask[(size_t)b * S_ + t0 + 16 + lr];

    // online softmax: per C-layout row r (actual row = r + 8*lh)
#pragma unroll
    for (int r = 0; r < 8; ++r) {
      float s0 = sc[0][r] * scale + mk0;
      float s1 = sc[1][r] * scale + mk1;
      float vmax = fmaxf(s0, s1);
#pragma unroll
      for (int off = 8; off >= 1; off >>= 1)
        vmax = fmaxf(vmax, __shfl_xor(vmax, off, 16));
      float mnew = fmaxf(mrun[r], vmax);
      float corr = __expf(mrun[r] - mnew);
      float p0 = __expf(s0 - mnew);
      float p1 = __expf(s1 - mnew);
      float ps = p0 + p1;
#pragma unroll
      for (int off = 8; off >= 1; off >>= 1)
        ps += __shfl_xor(ps, off, 16);
      lrun[r] = lrun[r] * corr + ps;
      mrun[r] = mnew;
#pragma unroll
      for (int nt = 0; nt < 4; ++nt) o[nt][r] *= corr;
      // stash P row-major (16x32) in per-wave LDS for C->A relayout
      int row = r + lh * 8;
      ldsP[wave][row * 32 + lr]      = (_Float16)p0;
      ldsP[wave][row * 32 + 16 + lr] = (_Float16)p1;
    }
    asm volatile("s_wait_dscnt 0" ::: "memory");

    // P as A fragment (16x32), V^T columns as B fragments
    v16h pa = *(const v16h*)(&ldsP[wave][lr * 32 + lh * 16]);
#pragma unroll
    for (int ds = 0; ds < 4; ++ds) {
      v16h vb = *(const v16h*)(&ldsVT[(ds * 16 + lr) * 32 + lh * 16]);
      o[ds] = __builtin_amdgcn_wmma_f32_16x16x32_f16(
          false, pa, false, vb, (short)0, o[ds], false, false);
    }
  }

  // Epilogue: O / l, fused head-merge store to [B,S,D] fp32
#pragma unroll
  for (int ds = 0; ds < 4; ++ds) {
    int d = ds * 16 + lr;
#pragma unroll
    for (int r = 0; r < 8; ++r) {
      int s = q0 + r + lh * 8;
      out[((size_t)b * S_ + s) * D_ + h * DH_ + d] = o[ds][r] / lrun[r];
    }
  }
}

// ---------------------------------------------------------------------------
extern "C" void kernel_launch(void* const* d_in, const int* in_sizes, int n_in,
                              void* d_out, int out_size, void* d_ws, size_t ws_size,
                              hipStream_t stream) {
  const float* x    = (const float*)d_in[0];
  const float* mask = (const float*)d_in[1];
  const float* W    = (const float*)d_in[2];
  const float* bias = (const float*)d_in[3];
  float* out = (float*)d_out;

  char* ws = (char*)d_ws;
  const size_t offWH = (size_t)B_ * S_ * D_ * 2;            // 12,582,912
  const size_t offQH = offWH + (size_t)D_ * D_ * 2;         // 13,762,560 (256-aligned)
  _Float16* xh = (_Float16*)ws;
  _Float16* wh = (_Float16*)(ws + offWH);
  _Float16* qh = (_Float16*)(ws + offQH);

  int total = B_ * S_ * D_;
  cvt_kernel<<<(total + 255) / 256, 256, 0, stream>>>(x, W, xh, wh);
  proj_kernel<<<dim3(B_ * S_ / 256, D_ / 64), 256, 0, stream>>>(xh, wh, bias, qh);
  attn_kernel<<<dim3(S_ / 128, B_ * H_), 256, 0, stream>>>(qh, mask, out);
}
